// transformer_82609400971585
// MI455X (gfx1250) — matrix operand order repaired and checked by reading, not compiled, not run on hardware
//
#include <hip/hip_runtime.h>
#include <math.h>

// ---------------------------------------------------------------------------
// Transformer (quirky) forward for MI455X / gfx1250, wave32 + WMMA bf16.
//
//  * softmax sits AFTER s@v  ->  O_h = Q_h (scale * K_h^T V_h): attention is
//    two tiny GEMM stages through a 64x64 per-head inner matrix.
//  * decoder "masked" attention runs on literal zero tokens -> one row + GEMV.
//  * GEMMs: v_wmma_f32_16x16x32_bf16, f32->bf16 (hardware packed convert)
//    staged into LDS pre-swizzled into the per-lane fragment layout
//    (ISA 7.12.2), register-prefetch double-buffered K loop (global loads
//    overlap WMMA via the split LOADcnt/DScnt counters).
//  * logits GEMM uses M-fastest grid order so each 2MB Wf panel is fetched
//    from HBM once and L2-hit by the other 7 M-tiles (Wf=524MB > 192MB L2).
// ---------------------------------------------------------------------------

#define Lq   1024
#define Dm   1024
#define Hh   16
#define NLAY 4
#define Vv   32000
#define KF   4096        // N_LAYERS * D
#define SCALE_QK 0.125f  // 1/sqrt(64)

typedef __attribute__((ext_vector_type(16))) __bf16 v16bf;
typedef __attribute__((ext_vector_type(8)))  __bf16 bf16x8;
typedef __attribute__((ext_vector_type(8)))  float  v8f;
typedef __attribute__((ext_vector_type(8)))  float  f32x8;

__device__ __forceinline__ float wred_sum(float v) {
  #pragma unroll
  for (int o = 16; o; o >>= 1) v += __shfl_xor(v, o, 32);
  return v;
}
__device__ __forceinline__ float wred_max(float v) {
  #pragma unroll
  for (int o = 16; o; o >>= 1) v = fmaxf(v, __shfl_xor(v, o, 32));
  return v;
}

// ---------------------------------------------------------------------------
// Tiled WMMA GEMM:  C[z] = scale * (A[z] @ B[z]^T) + bias[z] (opt relu) (+R[z])
//   A: [M,K] row-major (lda), B: [N,K] row-major (ldb), C: [M,N] (ldc).
//   8 waves / 256 threads; waves 4(M) x 2(N). Register-prefetch + LDS
//   ping-pong double buffering; one barrier per 32-K step.
//   gridSwap=1: M tiles on blockIdx.x (fast) for B-panel L2 reuse.
// ---------------------------------------------------------------------------
template<int BM, int BN>
__global__ __launch_bounds__(256)
void gemm_bt(const float* __restrict__ A, int lda, long aBatch,
             const float* __restrict__ B, int ldb, long bBatch,
             float*       __restrict__ C, int ldc, long cBatch,
             const float* __restrict__ bias, long biasBatch,
             const float* __restrict__ R, int ldr, long rBatch,
             int K, float scale, int relu, int gridSwap)
{
  constexpr int SM = BM / 64;        // 16x16 sub-tiles per wave (M)
  constexpr int SN = BN / 32;        // 16x16 sub-tiles per wave (N)
  constexpr int PA = (BM * 4) / 256; // A chunk iterations per thread
  constexpr int PB = (BN * 4) / 256; // B chunk iterations per thread
  static_assert(BM % 64 == 0 && BN % 32 == 0 && PA >= 1 && PB >= 1, "tile");

  // [buf][subtile][lane][16 bf16]: lane fragment = 32 contiguous bytes
  __shared__ unsigned short lA[2][BM / 16][32][16];
  __shared__ unsigned short lB[2][BN / 16][32][16];

  const int tid   = threadIdx.x;
  const int lane  = tid & 31;
  const int wave  = tid >> 5;
  const int waveM = wave & 3;
  const int waveN = wave >> 2;

  const int blockM = (gridSwap ? blockIdx.x : blockIdx.y) * BM;
  const int blockN = (gridSwap ? blockIdx.y : blockIdx.x) * BN;
  const int z = blockIdx.z;
  A += (size_t)z * aBatch;
  B += (size_t)z * bBatch;
  C += (size_t)z * cBatch;
  if (bias) bias += (size_t)z * biasBatch;
  if (R)    R    += (size_t)z * rBatch;

  v8f acc[SM][SN];
  #pragma unroll
  for (int i = 0; i < SM; i++)
    #pragma unroll
    for (int j = 0; j < SN; j++)
      #pragma unroll
      for (int v = 0; v < 8; v++) acc[i][j][v] = 0.0f;

  float4 ra[PA][2], rb[PB][2];

  auto prefetch = [&](int k0) {
    #pragma unroll
    for (int it = 0; it < PA; it++) {
      const int c = tid + 256 * it, m = c >> 2, q = c & 3;
      const float* src = A + (size_t)(blockM + m) * lda + k0 + q * 8;
      ra[it][0] = ((const float4*)src)[0];
      ra[it][1] = ((const float4*)src)[1];
    }
    #pragma unroll
    for (int it = 0; it < PB; it++) {
      const int c = tid + 256 * it, n = c >> 2, q = c & 3;
      const float* src = B + (size_t)(blockN + n) * ldb + k0 + q * 8;
      rb[it][0] = ((const float4*)src)[0];
      rb[it][1] = ((const float4*)src)[1];
    }
  };
  auto cvt8 = [](float4 a, float4 b) -> bf16x8 {
    f32x8 f;
    f[0] = a.x; f[1] = a.y; f[2] = a.z; f[3] = a.w;
    f[4] = b.x; f[5] = b.y; f[6] = b.z; f[7] = b.w;
    return __builtin_convertvector(f, bf16x8);   // hw packed f32->bf16 (RNE)
  };
  auto stage = [&](int buf) {
    // A-layout (ISA 7.12.2): lane-half = chunk&1, 16B-quarter = chunk>>1
    #pragma unroll
    for (int it = 0; it < PA; it++) {
      const int c = tid + 256 * it, m = c >> 2, q = c & 3;
      *(bf16x8*)&lA[buf][m >> 4][(m & 15) + ((q & 1) << 4)][(q >> 1) * 8] =
          cvt8(ra[it][0], ra[it][1]);
    }
    // B-layout (same as A): lane-half = chunk&1, 16B-quarter = chunk>>1
    #pragma unroll
    for (int it = 0; it < PB; it++) {
      const int c = tid + 256 * it, n = c >> 2, q = c & 3;
      *(bf16x8*)&lB[buf][n >> 4][(n & 15) + ((q & 1) << 4)][(q >> 1) * 8] =
          cvt8(rb[it][0], rb[it][1]);
    }
  };

  prefetch(0);
  int buf = 0;
  for (int k0 = 0; k0 < K; k0 += 32) {
    stage(buf);
    __syncthreads();
    if (k0 + 32 < K) prefetch(k0 + 32);   // in flight during WMMA below

    v16bf af[SM], bfv[SN];
    #pragma unroll
    for (int i = 0; i < SM; i++)
      af[i] = *(const v16bf*)&lA[buf][waveM * SM + i][lane][0];
    #pragma unroll
    for (int j = 0; j < SN; j++)
      bfv[j] = *(const v16bf*)&lB[buf][waveN * SN + j][lane][0];
    #pragma unroll
    for (int i = 0; i < SM; i++)
      #pragma unroll
      for (int j = 0; j < SN; j++)
        acc[i][j] = __builtin_amdgcn_wmma_f32_16x16x32_bf16(
            false, af[i], false, bfv[j], (short)0, acc[i][j], false, false);
    buf ^= 1;
  }

  // epilogue: C/D layout: lane<16 -> rows v, lane>=16 -> rows v+8; col=lane&15
  const int radd = (lane >> 4) * 8;
  const int nl   = lane & 15;
  #pragma unroll
  for (int j = 0; j < SN; j++) {
    const int col = blockN + waveN * (BN / 2) + j * 16 + nl;
    const float bb = bias ? bias[col] : 0.0f;
    #pragma unroll
    for (int i = 0; i < SM; i++) {
      const int rowBase = blockM + waveM * (BM / 4) + i * 16 + radd;
      #pragma unroll
      for (int v = 0; v < 8; v++) {
        float val = acc[i][j][v] * scale + bb;
        if (relu) val = fmaxf(val, 0.0f);
        const int r = rowBase + v;
        if (R) val += R[(size_t)r * ldr + col];
        C[(size_t)r * ldc + col] = val;
      }
    }
  }
}

// ---------------------------------------------------------------------------
__global__ void transpose_f32(const float* __restrict__ in,
                              float* __restrict__ out, int rows, int cols) {
  __shared__ float t[32][33];
  const int bc = blockIdx.x * 32, br = blockIdx.y * 32;
  #pragma unroll
  for (int j = threadIdx.y; j < 32; j += 8)
    t[j][threadIdx.x] = in[(size_t)(br + j) * cols + bc + threadIdx.x];
  __syncthreads();
  #pragma unroll
  for (int j = threadIdx.y; j < 32; j += 8)
    out[(size_t)(bc + j) * rows + br + threadIdx.x] = t[threadIdx.x][j];
}

// softmax over dv=64 per (l,head) segment; 8 waves/block, 1 segment/wave
__global__ void softmax_seg(float* __restrict__ X) {
  const int seg = blockIdx.x * 8 + (threadIdx.x >> 5);
  const int ln  = threadIdx.x & 31;
  float* p = X + (size_t)seg * 64;
  float a = p[ln], b = p[ln + 32];
  const float m = wred_max(fmaxf(a, b));
  float e0 = expf(a - m), e1 = expf(b - m);
  const float inv = 1.0f / wred_sum(e0 + e1);
  p[ln] = e0 * inv; p[ln + 32] = e1 * inv;
}

// ---- global LayerNorm over all L*D elements (deterministic 2-stage) -------
__global__ void row_stats(const float* __restrict__ X, float2* __restrict__ part) {
  const int r = blockIdx.x;
  const float* row = X + (size_t)r * Dm;
  float s = 0.f, ss = 0.f;
  for (int c = threadIdx.x; c < Dm; c += 256) { float v = row[c]; s += v; ss += v * v; }
  __shared__ float as[8], bs[8];
  s = wred_sum(s); ss = wred_sum(ss);
  if ((threadIdx.x & 31) == 0) { as[threadIdx.x >> 5] = s; bs[threadIdx.x >> 5] = ss; }
  __syncthreads();
  if (threadIdx.x == 0) {
    float S = 0.f, SS = 0.f;
    for (int i = 0; i < 8; i++) { S += as[i]; SS += bs[i]; }
    part[r] = make_float2(S, SS);
  }
}

__global__ void stats_final(const float2* __restrict__ part, float* __restrict__ ms,
                            int n, float inv) {
  float s = 0.f, ss = 0.f;
  for (int i = threadIdx.x; i < n; i += 256) { float2 p = part[i]; s += p.x; ss += p.y; }
  __shared__ float as[8], bs[8];
  s = wred_sum(s); ss = wred_sum(ss);
  if ((threadIdx.x & 31) == 0) { as[threadIdx.x >> 5] = s; bs[threadIdx.x >> 5] = ss; }
  __syncthreads();
  if (threadIdx.x == 0) {
    float S = 0.f, SS = 0.f;
    for (int i = 0; i < 8; i++) { S += as[i]; SS += bs[i]; }
    const float mu = S * inv;
    ms[0] = mu; ms[1] = rsqrtf(SS * inv - mu * mu + 1e-5f);
  }
}

__global__ void ln_apply(const float* __restrict__ X, const float* __restrict__ ms,
                         const float* __restrict__ g, const float* __restrict__ b,
                         float* __restrict__ Y, int ldy) {
  const size_t i = (size_t)blockIdx.x * 256 + threadIdx.x;
  const float mu = ms[0], ri = ms[1];
  Y[(i >> 10) * (size_t)ldy + (i & 1023)] = (X[i] - mu) * ri * g[i] + b[i];
}

__global__ void vec_stats(const float* __restrict__ x, int n, float* __restrict__ ms,
                          float inv) {
  float s = 0.f, ss = 0.f;
  for (int i = threadIdx.x; i < n; i += 256) { float v = x[i]; s += v; ss += v * v; }
  __shared__ float as[8], bs[8];
  s = wred_sum(s); ss = wred_sum(ss);
  if ((threadIdx.x & 31) == 0) { as[threadIdx.x >> 5] = s; bs[threadIdx.x >> 5] = ss; }
  __syncthreads();
  if (threadIdx.x == 0) {
    float S = 0.f, SS = 0.f;
    for (int i = 0; i < 8; i++) { S += as[i]; SS += bs[i]; }
    const float mu = S * inv;
    ms[0] = mu; ms[1] = rsqrtf(SS * inv - mu * mu + 1e-5f);
  }
}

__global__ void bc_ln_apply(const float* __restrict__ rowv, const float* __restrict__ ms,
                            const float* __restrict__ g, const float* __restrict__ b,
                            float* __restrict__ Y) {
  const size_t i = (size_t)blockIdx.x * 256 + threadIdx.x;
  Y[i] = (rowv[i & 1023] - ms[0]) * ms[1] * g[i] + b[i];
}

// decoder masked attention on zero tokens: one wave per head
__global__ void dec_const_row(const float* __restrict__ bq, const float* __restrict__ bk,
                              const float* __restrict__ bv, float* __restrict__ rowo) {
  const int h  = blockIdx.x * 8 + (threadIdx.x >> 5);
  const int ln = threadIdx.x & 31;
  const float* q = bq + h * 64; const float* k = bk + h * 64; const float* v = bv + h * 64;
  const float c = wred_sum(q[ln] * k[ln] + q[ln + 32] * k[ln + 32]) * SCALE_QK;
  float o0 = (float)Lq * c * v[ln], o1 = (float)Lq * c * v[ln + 32];
  const float m = wred_max(fmaxf(o0, o1));
  float e0 = expf(o0 - m), e1 = expf(o1 - m);
  const float inv = 1.0f / wred_sum(e0 + e1);
  rowo[h * 64 + ln] = e0 * inv; rowo[h * 64 + ln + 32] = e1 * inv;
}

// out[n] = dot(xrow, W[n,:]) + bias[n]; one wave per output
__global__ void gemv_bt(const float* __restrict__ xrow, const float* __restrict__ W,
                        const float* __restrict__ bias, float* __restrict__ out) {
  const int n  = blockIdx.x * 8 + (threadIdx.x >> 5);
  const int ln = threadIdx.x & 31;
  const float* w = W + (size_t)n * Dm;
  float s = 0.f;
  for (int k = ln; k < Dm; k += 32) s += xrow[k] * w[k];
  s = wred_sum(s);
  if (ln == 0) out[n] = s + bias[n];
}

// softmax over V per row (final output), in-place
__global__ void softmax_rows(float* __restrict__ X, int n) {
  float* row = X + (size_t)blockIdx.x * n;
  __shared__ float red[8];
  const int t = threadIdx.x, wv = t >> 5, ln = t & 31;
  float m = -INFINITY;
  for (int i = t; i < n; i += 256) m = fmaxf(m, row[i]);
  m = wred_max(m);
  if (ln == 0) red[wv] = m;
  __syncthreads();
  float mm = red[0];
  for (int i = 1; i < 8; i++) mm = fmaxf(mm, red[i]);
  __syncthreads();
  float s = 0.f;
  for (int i = t; i < n; i += 256) { float e = expf(row[i] - mm); row[i] = e; s += e; }
  s = wred_sum(s);
  if (ln == 0) red[wv] = s;
  __syncthreads();
  float S = 0.f;
  for (int i = 0; i < 8; i++) S += red[i];
  const float inv = 1.0f / S;
  for (int i = t; i < n; i += 256) row[i] *= inv;
}

// ---------------------------------------------------------------------------
extern "C" void kernel_launch(void* const* d_in, const int* in_sizes, int n_in,
                              void* d_out, int out_size, void* d_ws, size_t ws_size,
                              hipStream_t stream) {
  (void)in_sizes; (void)n_in; (void)out_size; (void)ws_size;
  auto F = [&](int i) { return (const float*)d_in[i]; };
  const float *x    = F(0);
  const float *Wq_e = F(1),  *bq_e = F(2),  *Wk_e = F(3),  *bk_e = F(4);
  const float *Wv_e = F(5),  *bv_e = F(6),  *Wo_e = F(7),  *bo_e = F(8);
  const float *bq_m = F(10), *bk_m = F(12), *bv_m = F(14);
  const float *Wo_m = F(15), *bo_m = F(16);
  const float *Wq_c = F(17), *bq_c = F(18), *Wk_c = F(19), *bk_c = F(20);
  const float *Wv_c = F(21), *bv_c = F(22), *Wo_c = F(23), *bo_c = F(24);
  const float *e1w = F(25), *e1b = F(26), *e2w = F(27), *e2b = F(28);
  const float *d1w = F(29), *d1b = F(30), *d2w = F(31), *d2b = F(32);
  const float *ln_g = F(33), *ln_b = F(34), *Wf = F(35), *bff = F(36);

  // workspace layout (floats); requires ~61 MB
  float* W0 = (float*)d_ws;
  const size_t M1 = (size_t)Lq * Dm;
  float* Qb   = W0 + 0 * M1;
  float* Kb   = W0 + 1 * M1;
  float* Vb   = W0 + 2 * M1;
  float* Ob   = W0 + 3 * M1;
  float* Kt   = W0 + 4 * M1;
  float* Vt   = W0 + 5 * M1;
  float* F1   = W0 + 6 * M1;
  float* Yb   = W0 + 7 * M1;
  float* OutB = W0 + 8 * M1;
  float* Dbuf = W0 + 9 * M1;
  float* Out2 = W0 + 10 * M1;
  float* OUTS = W0 + 11 * M1;            // [L, 4*D]
  float* Tt   = W0 + 15 * M1;            // [H,64,64] per-head scale*(K^T V)^T
  float* rowo = Tt + Hh * 64 * 64;
  float* rowd = rowo + Dm;
  float* ms   = rowd + Dm;
  float2* part = (float2*)(ms + 16);

  const float invLD = 1.0f / ((float)Lq * (float)Dm);

  auto G128 = [&](const float* A, int lda, const float* B, int ldb, float* C, int ldc,
                  const float* bias, const float* R, int ldr, int M, int N, int K,
                  float sc, int relu) {
    gemm_bt<128, 128><<<dim3(N / 128, M / 128, 1), 256, 0, stream>>>(
        A, lda, 0, B, ldb, 0, C, ldc, 0, bias, 0, R, ldr, 0, K, sc, relu, 0);
  };
  auto LN = [&](const float* Xin, float* Yout, int ldy) {
    row_stats<<<Lq, 256, 0, stream>>>(Xin, part);
    stats_final<<<1, 256, 0, stream>>>(part, ms, Lq, invLD);
    ln_apply<<<(Lq * Dm) / 256, 256, 0, stream>>>(Xin, ms, ln_g, ln_b, Yout, ldy);
  };
  // O = softmax_dv( Q (scale K^T V) ); out = LN( O Wo^T + bo + resid )
  auto MHA = [&](const float* xq, const float* xkv,
                 const float* Wq, const float* bq, const float* Wk, const float* bk,
                 const float* Wv, const float* bv, const float* Wo, const float* bo,
                 const float* resid, float* outBuf) {
    G128(xq,  Dm, Wq, Dm, Qb, Dm, bq, nullptr, 0, Lq, Dm, Dm, 1.f, 0);
    G128(xkv, Dm, Wk, Dm, Kb, Dm, bk, nullptr, 0, Lq, Dm, Dm, 1.f, 0);
    G128(xkv, Dm, Wv, Dm, Vb, Dm, bv, nullptr, 0, Lq, Dm, Dm, 1.f, 0);
    transpose_f32<<<dim3(Dm / 32, Lq / 32), dim3(32, 8), 0, stream>>>(Kb, Kt, Lq, Dm);
    transpose_f32<<<dim3(Dm / 32, Lq / 32), dim3(32, 8), 0, stream>>>(Vb, Vt, Lq, Dm);
    // Tt[h][v][k] = scale * sum_l V[l,hv]*K[l,hk]   (batched over heads)
    gemm_bt<64, 64><<<dim3(1, 1, Hh), 256, 0, stream>>>(
        Vt, Lq, 64L * Lq, Kt, Lq, 64L * Lq, Tt, 64, 64L * 64,
        nullptr, 0, nullptr, 0, 0, Lq, SCALE_QK, 0, 0);
    // O[l, h*64+v] = sum_k Q[l,h*64+k] * Tt[h][v][k]
    gemm_bt<128, 64><<<dim3(1, Lq / 128, Hh), 256, 0, stream>>>(
        Qb, Dm, 64L, Tt, 64, 64L * 64, Ob, Dm, 64L,
        nullptr, 0, nullptr, 0, 0, 64, 1.f, 0, 0);
    softmax_seg<<<(Lq * Hh) / 8, 256, 0, stream>>>(Ob);
    G128(Ob, Dm, Wo, Dm, Yb, Dm, bo, resid, Dm, Lq, Dm, Dm, 1.f, 0);
    LN(Yb, outBuf, Dm);
  };
  auto FFN = [&](const float* xin, const float* w1, const float* b1,
                 const float* w2, const float* b2, float* outBuf, int ldOut) {
    G128(xin, Dm, w1, Dm, F1, Dm, b1, nullptr, 0, Lq, Dm, Dm, 1.f, 1);
    G128(F1, Dm, w2, Dm, Yb, Dm, b2, xin, Dm, Lq, Dm, Dm, 1.f, 1); // relu then +resid
    LN(Yb, outBuf, ldOut);
  };

  const float* identity = x;
  for (int l = 0; l < NLAY; l++) {
    const size_t wOff = (size_t)l * Dm * Dm, bOff = (size_t)l * Dm;
    // encoder: attention always on ORIGINAL x
    MHA(x, x, Wq_e + wOff, bq_e + bOff, Wk_e + wOff, bk_e + bOff,
        Wv_e + wOff, bv_e + bOff, Wo_e + wOff, bo_e + bOff, identity, OutB);
    FFN(OutB, e1w + wOff, e1b + bOff, e2w + wOff, e2b + bOff, OutB, Dm);
    // decoder masked attention on zero start tokens (row-constant pre-LN)
    dec_const_row<<<Hh / 8, 256, 0, stream>>>(bq_m + bOff, bk_m + bOff, bv_m + bOff, rowo);
    gemv_bt<<<Dm / 8, 256, 0, stream>>>(rowo, Wo_m + wOff, bo_m + bOff, rowd);
    vec_stats<<<1, 256, 0, stream>>>(rowd, Dm, ms, 1.0f / (float)Dm);
    bc_ln_apply<<<(Lq * Dm) / 256, 256, 0, stream>>>(rowd, ms, ln_g, ln_b, Dbuf);
    // cross attention: q from d, k/v from encoder out; residual = d
    MHA(Dbuf, OutB, Wq_c + wOff, bq_c + bOff, Wk_c + wOff, bk_c + bOff,
        Wv_c + wOff, bv_c + bOff, Wo_c + wOff, bo_c + bOff, Dbuf, Out2);
    // decoder FFN -> slot l of concat buffer
    FFN(Out2, d1w + wOff, d1b + bOff, d2w + wOff, d2b + bOff, OUTS + (size_t)l * Dm, KF);
    identity = Out2;  // reference: identity is the cross-attn LN output
  }

  // logits = OUTS @ Wf^T + bf; gridSwap=1 -> M tiles fastest so the 8 blocks
  // sharing a Wf N-panel are launch-adjacent (panel read from HBM once, L2 hit
  // for the rest). Then softmax over V per row.
  float* logits = (float*)d_out;
  gemm_bt<128, 128><<<dim3(Lq / 128, Vv / 128, 1), 256, 0, stream>>>(
      OUTS, KF, 0, Wf, KF, 0, logits, Vv, 0, bff, 0, nullptr, 0, 0, KF, 1.f, 0, 1);
  softmax_rows<<<Lq, 256, 0, stream>>>(logits, Vv);
}
